// FrameSimilarity_81277961109919
// MI455X (gfx1250) — matrix operand order repaired and checked by reading, not compiled, not run on hardware
//
#include <hip/hip_runtime.h>

typedef __attribute__((ext_vector_type(16))) __bf16 v16bf;
typedef __attribute__((ext_vector_type(8)))  __bf16 v8bf;
typedef __attribute__((ext_vector_type(8)))  float  v8f;

namespace {
constexpr int kB      = 4;
constexpr int kC1     = 1024;
constexpr int kC2     = 1024;
constexpr int kT      = 2048;
constexpr int kInCh   = 2048;   // C1 + C2
constexpr int kSim    = 128;
constexpr int kLookup = 101;
constexpr int kOut    = 128;
constexpr int kHW     = 16;
constexpr int kTok    = kB * kT;        // 8192 tokens total
constexpr int kKT1    = kInCh / 32;     // 64 K-tiles, projection GEMM
constexpr int kNT1    = kSim / 16;      // 8  N-tiles
constexpr int kKT2    = 128 / 32;       // 4  K-tiles, FC GEMM (101 padded to 128)
constexpr int kNT2    = kOut / 16;      // 8
constexpr int kBandPad = 136;           // 128 + 8 pad: 16B-aligned rows, no LDS bank conflicts
constexpr int kHalf   = (kLookup - 1) / 2;  // 50
}

__device__ __forceinline__ v16bf combine8(v8bf lo, v8bf hi) {
  v16bf r;
#pragma unroll
  for (int i = 0; i < 8; ++i) { r[i] = lo[i]; r[i + 8] = hi[i]; }
  return r;
}

// ---------------------------------------------------------------------------
// Kernel 1: mean-pool over H*W=16 and transpose (B,C,T) -> (B*T, C) in bf16.
// Grid: (T/32, IN_CH/32, B), block 256. Reads coalesced over t (2KB/wave),
// writes token-major via LDS transpose. This is the HBM-bound pass (1.07 GB).
// ---------------------------------------------------------------------------
__global__ void pool_kernel(const float* __restrict__ x1,
                            const float* __restrict__ x2,
                            __bf16* __restrict__ pooled) {
  __shared__ __bf16 tile[32][33];
  const int t0 = blockIdx.x * 32;
  const int c0 = blockIdx.y * 32;
  const int b  = blockIdx.z;

  for (int p = threadIdx.x; p < 32 * 32; p += 256) {
    const int tl = p & 31;          // t fastest -> contiguous 64B per thread
    const int cl = p >> 5;
    const int c  = c0 + cl;
    const int t  = t0 + tl;
    const float* src = (c < kC1)
        ? (x1 + ((size_t)(b * kC1 + c)         * kT + t) * kHW)
        : (x2 + ((size_t)(b * kC2 + (c - kC1)) * kT + t) * kHW);
    const float4* s4 = reinterpret_cast<const float4*>(src);
    float4 f0 = s4[0], f1 = s4[1], f2 = s4[2], f3 = s4[3];
    float s = ((f0.x + f0.y) + (f0.z + f0.w)) + ((f1.x + f1.y) + (f1.z + f1.w)) +
              ((f2.x + f2.y) + (f2.z + f2.w)) + ((f3.x + f3.y) + (f3.z + f3.w));
    tile[tl][cl] = (__bf16)(s * (1.0f / 16.0f));
  }
  __syncthreads();
  for (int p = threadIdx.x; p < 32 * 32; p += 256) {
    const int cl = p & 31;          // c fastest -> contiguous writes
    const int tl = p >> 5;
    pooled[((size_t)(b * kT + t0 + tl)) * kInCh + (c0 + cl)] = tile[tl][cl];
  }
}

// ---------------------------------------------------------------------------
// Kernel 2: pack W_proj (2048x128 f32, KxN row-major) into WMMA B-fragment
// layout [kt][nt][lane][e] bf16:  n = nt*16 + (lane&15), k = kt*32 + (lane&16) + e
// ---------------------------------------------------------------------------
__global__ void pack_wproj(const float* __restrict__ W, __bf16* __restrict__ Wp) {
  const int idx = blockIdx.x * 256 + threadIdx.x;
  if (idx >= kKT1 * kNT1 * 32 * 16) return;
  const int e    = idx & 15;
  const int lane = (idx >> 4) & 31;
  const int nt   = (idx >> 9) & 7;
  const int kt   = idx >> 12;
  const int n = nt * 16 + (lane & 15);
  const int k = kt * 32 + (lane & 16) + e;
  Wp[idx] = (__bf16)W[(size_t)k * kSim + n];
}

// Kernel 2b: same for W_fc (101x128), K zero-padded to 128.
__global__ void pack_wfc(const float* __restrict__ W, __bf16* __restrict__ Wp) {
  const int idx = blockIdx.x * 256 + threadIdx.x;
  if (idx >= kKT2 * kNT2 * 32 * 16) return;
  const int e    = idx & 15;
  const int lane = (idx >> 4) & 31;
  const int nt   = (idx >> 9) & 7;
  const int kt   = idx >> 12;
  const int n = nt * 16 + (lane & 15);
  const int k = kt * 32 + (lane & 16) + e;
  Wp[idx] = (k < kLookup) ? (__bf16)W[(size_t)k * kOut + n] : (__bf16)0.0f;
}

// ---------------------------------------------------------------------------
// Kernel 3: projection GEMM (8192x2048)x(2048x128) via WMMA bf16 + bias +
// in-register row L2-norm (shfl_xor within the 16-lane half-wave).
// Explicit 2-deep register double buffering: prefetch kt+1's A fragment and
// all 8 B fragments while the current 8 WMMAs execute, so the inner loop is
// not serialized on L2 latency (only ~512 waves in flight for this kernel).
// ---------------------------------------------------------------------------
__global__ void proj_kernel(const __bf16* __restrict__ Apool,
                            const __bf16* __restrict__ Wp,
                            const float* __restrict__ bias,
                            __bf16* __restrict__ Pn) {
  const int wave  = threadIdx.x >> 5;
  const int lane  = threadIdx.x & 31;
  const int mtile = blockIdx.x * 4 + wave;        // 0..511
  const int ln    = lane & 15;
  const int hi    = lane >> 4;
  const int tok_a = (mtile << 4) + ln;            // A-matrix row for this lane

  const __bf16* arow  = Apool + (size_t)tok_a * kInCh + hi * 16;
  const __bf16* bbase = Wp + (size_t)lane * 16;

  auto loadA = [&](int kt) -> v16bf {
    v8bf a0 = *(const v8bf*)(arow + kt * 32);
    v8bf a1 = *(const v8bf*)(arow + kt * 32 + 8);
    return combine8(a0, a1);
  };

  v8f acc[kNT1] = {};
  v16bf aA, aB, bA[kNT1], bB[kNT1];

  aA = loadA(0);
#pragma unroll
  for (int nt = 0; nt < kNT1; ++nt)
    bA[nt] = *(const v16bf*)(bbase + nt * 512);

  for (int kt = 0; kt < kKT1; kt += 2) {
    const int k1 = kt + 1;                              // kKT1 is even
    aB = loadA(k1);
#pragma unroll
    for (int nt = 0; nt < kNT1; ++nt)
      bB[nt] = *(const v16bf*)(bbase + (size_t)k1 * 4096 + nt * 512);
#pragma unroll
    for (int nt = 0; nt < kNT1; ++nt)
      acc[nt] = __builtin_amdgcn_wmma_f32_16x16x32_bf16(
          false, aA, false, bA[nt], (short)0, acc[nt], false, false);

    const int k2 = (kt + 2 < kKT1) ? kt + 2 : kt;       // dummy reload last iter
    aA = loadA(k2);
#pragma unroll
    for (int nt = 0; nt < kNT1; ++nt)
      bA[nt] = *(const v16bf*)(bbase + (size_t)k2 * 4096 + nt * 512);
#pragma unroll
    for (int nt = 0; nt < kNT1; ++nt)
      acc[nt] = __builtin_amdgcn_wmma_f32_16x16x32_bf16(
          false, aB, false, bB[nt], (short)0, acc[nt], false, false);
  }

  // bias add (D layout: col = nt*16 + ln, row = v + 8*hi)
#pragma unroll
  for (int nt = 0; nt < kNT1; ++nt) {
    const float bv = bias[nt * 16 + ln];
#pragma unroll
    for (int v = 0; v < 8; ++v) acc[nt][v] += bv;
  }

  // Row sum-of-squares: reduce over the 16 lanes of this half-wave + 8 n-tiles.
  float inv[8];
#pragma unroll
  for (int v = 0; v < 8; ++v) {
    float s = 0.0f;
#pragma unroll
    for (int nt = 0; nt < kNT1; ++nt) s += acc[nt][v] * acc[nt][v];
    for (int m = 1; m < 16; m <<= 1) s += __shfl_xor(s, m, 32);  // stays in half
    const float nrm = sqrtf(s);
    inv[v] = 1.0f / fmaxf(nrm, 1e-12f);
  }

#pragma unroll
  for (int nt = 0; nt < kNT1; ++nt) {
#pragma unroll
    for (int v = 0; v < 8; ++v) {
      const int tok = (mtile << 4) + v + hi * 8;
      const int col = nt * 16 + ln;
      Pn[(size_t)tok * kSim + col] = (__bf16)(acc[nt][v] * inv[v]);
    }
  }
}

// ---------------------------------------------------------------------------
// Kernel 4: banded cosine sims (|s-t|<=50) via WMMA, scattered into an LDS
// band tile (16 x 128, zero padded past 101), then band @ W_fc + bias, ReLU.
// Both WMMA loops use 2-deep register double buffering as in kernel 3.
// ---------------------------------------------------------------------------
__global__ void band_fc_kernel(const __bf16* __restrict__ Pn,
                               const __bf16* __restrict__ Wfc,
                               const float* __restrict__ bfc,
                               float* __restrict__ out) {
  __shared__ __bf16 band[4][16][kBandPad];
  const int wave = threadIdx.x >> 5;
  const int lane = threadIdx.x & 31;
  const int grp  = blockIdx.x * 4 + wave;   // 0..511
  const int b    = grp >> 7;                // 128 token-groups per batch
  const int t0   = (grp & 127) << 4;
  const int ln   = lane & 15;
  const int hi   = lane >> 4;

  // zero band tile
  __bf16* bnd = &band[wave][0][0];
  for (int i = lane; i < 16 * kBandPad; i += 32) bnd[i] = (__bf16)0.0f;

  // A fragments: p-hat rows t0..t0+15, K = 128 -> 4 k-tiles (kept in regs)
  const __bf16* Prow = Pn + ((size_t)(b * kT) + t0 + ln) * kSim + hi * 16;
  v16bf afrag[kKT2];
#pragma unroll
  for (int kt = 0; kt < kKT2; ++kt) {
    v8bf a0 = *(const v8bf*)(Prow + kt * 32);
    v8bf a1 = *(const v8bf*)(Prow + kt * 32 + 8);
    afrag[kt] = combine8(a0, a1);
  }

  const v16bf bz = {};
  // B-fragment loader for sims s-tile j: n = ln -> s = s0+ln, contiguous 32B row reads
  auto load_s = [&](int j, v16bf* dst, bool& okOut, int& s0Out) {
    const int s0   = t0 - kHalf + j * 16;
    const int srow = s0 + ln;
    const bool ok  = (srow >= 0) && (srow < kT);
    const __bf16* Brow =
        Pn + ((size_t)(b * kT) + (ok ? srow : 0)) * kSim + hi * 16;
#pragma unroll
    for (int kt = 0; kt < kKT2; ++kt) {
      v16bf f = *(const v16bf*)(Brow + kt * 32);
      dst[kt] = ok ? f : bz;
    }
    okOut = ok; s0Out = s0;
  };
  // 4 WMMAs + scatter D into band: t = t0 + v + 8*hi, s = s0 + ln, w = s-t+50
  auto sim_tile = [&](const v16bf* bf, bool ok, int s0) {
    v8f acc = {};
#pragma unroll
    for (int kt = 0; kt < kKT2; ++kt)
      acc = __builtin_amdgcn_wmma_f32_16x16x32_bf16(
          false, afrag[kt], false, bf[kt], (short)0, acc, false, false);
#pragma unroll
    for (int v = 0; v < 8; ++v) {
      const int tl = v + hi * 8;
      const int w  = (s0 + ln) - (t0 + tl) + kHalf;
      if (ok && w >= 0 && w < kLookup) band[wave][tl][w] = (__bf16)acc[v];
    }
  };

  v16bf sA[kKT2], sB[kKT2];
  bool okA, okB; int s0A, s0B;
  load_s(0, sA, okA, s0A);
  for (int j = 0; j < 8; j += 2) {               // 8 s-tiles cover [t0-50, t0+65]
    load_s(j + 1, sB, okB, s0B);
    sim_tile(sA, okA, s0A);
    load_s((j + 2 < 8) ? j + 2 : j, sA, okA, s0A);
    sim_tile(sB, okB, s0B);
  }
  __syncthreads();

  // FC: band(16x128) @ Wfc(128x128), A from LDS
  const __bf16* bl = &band[wave][ln][hi * 16];
  v16bf afc[kKT2];
#pragma unroll
  for (int kt = 0; kt < kKT2; ++kt) {
    v8bf a0 = *(const v8bf*)(bl + kt * 32);
    v8bf a1 = *(const v8bf*)(bl + kt * 32 + 8);
    afc[kt] = combine8(a0, a1);
  }

  auto load_w = [&](int nt, v16bf* d) {
#pragma unroll
    for (int kt = 0; kt < kKT2; ++kt)
      d[kt] = *(const v16bf*)(Wfc + (size_t)kt * (kNT2 * 512) + nt * 512 +
                              lane * 16);
  };
  auto fc_tile = [&](const v16bf* w, int nt) {
    v8f acc = {};
#pragma unroll
    for (int kt = 0; kt < kKT2; ++kt)
      acc = __builtin_amdgcn_wmma_f32_16x16x32_bf16(
          false, afc[kt], false, w[kt], (short)0, acc, false, false);
    const float bv = bfc[nt * 16 + ln];
#pragma unroll
    for (int v = 0; v < 8; ++v) {
      const int t   = t0 + v + hi * 8;
      const int col = nt * 16 + ln;
      out[((size_t)(b * kT) + t) * kOut + col] = fmaxf(acc[v] + bv, 0.0f);
    }
  };

  v16bf wA[kKT2], wB[kKT2];
  load_w(0, wA);
  for (int nt = 0; nt < kNT2; nt += 2) {
    load_w(nt + 1, wB);
    fc_tile(wA, nt);
    load_w((nt + 2 < kNT2) ? nt + 2 : nt, wA);
    fc_tile(wB, nt + 1);
  }
}

// ---------------------------------------------------------------------------
extern "C" void kernel_launch(void* const* d_in, const int* in_sizes, int n_in,
                              void* d_out, int out_size, void* d_ws, size_t ws_size,
                              hipStream_t stream) {
  const float* x1    = (const float*)d_in[0];
  const float* x2    = (const float*)d_in[1];
  const float* Wproj = (const float*)d_in[2];
  const float* bproj = (const float*)d_in[3];
  const float* Wfc   = (const float*)d_in[4];
  const float* bfc   = (const float*)d_in[5];
  float* out = (float*)d_out;

  char* ws = (char*)d_ws;
  size_t off = 0;
  __bf16* pooled = (__bf16*)(ws + off); off += (size_t)kTok * kInCh * 2;   // 33.5 MB
  __bf16* Pn     = (__bf16*)(ws + off); off += (size_t)kTok * kSim * 2;    // 2 MB
  __bf16* WpP    = (__bf16*)(ws + off); off += (size_t)kKT1 * kNT1 * 512 * 2;
  __bf16* WpF    = (__bf16*)(ws + off); off += (size_t)kKT2 * kNT2 * 512 * 2;
  (void)in_sizes; (void)n_in; (void)out_size; (void)ws_size;

  pool_kernel<<<dim3(kT / 32, kInCh / 32, kB), 256, 0, stream>>>(x1, x2, pooled);
  pack_wproj<<<(kKT1 * kNT1 * 512 + 255) / 256, 256, 0, stream>>>(Wproj, WpP);
  pack_wfc<<<(kKT2 * kNT2 * 512 + 255) / 256, 256, 0, stream>>>(Wfc, WpF);
  proj_kernel<<<kTok / 16 / 4, 128, 0, stream>>>(pooled, WpP, bproj, Pn);
  band_fc_kernel<<<kTok / 16 / 4, 128, 0, stream>>>(Pn, WpF, bfc, out);
}
